// ConversationLSTM_4234837754539
// MI455X (gfx1250) — compile-verified, matrix operand order checked by reading, not run on hardware
//
#include <hip/hip_runtime.h>
#include <hip/hip_bf16.h>
#include <cstdint>
#include <cstddef>

typedef __attribute__((ext_vector_type(16))) _Float16 v16h;
typedef __attribute__((ext_vector_type(8)))  float    v8f;
typedef int v4i_t __attribute__((vector_size(16)));

#define ASG __attribute__((address_space(1)))
#define ASL __attribute__((address_space(3)))

#if __has_builtin(__builtin_amdgcn_global_load_async_to_lds_b128) && \
    __has_builtin(__builtin_amdgcn_s_wait_asynccnt)
#define USE_ASYNC_LDS 1
#else
#define USE_ASYNC_LDS 0
#endif

// ---------------- problem dims ----------------
#define Bq    256
#define Tq    256
#define Hq    512
#define Lq    3
#define G4H   2048          // 4*H
#define KCAT  1024          // H (x-part) + H (h-part)
#define NSEQ  3
#define HDIM2 1024          // 2*H
#define HDIM3 102           // H//5
#define R3H   1536          // 3*H

// ---------------- LSTM GEMM tiling ----------------
#define BM 128
#define BN 128
#define KSTG 128                               // K per staged A tile (4 WMMA-K chunks)
#define NSTG (KCAT / KSTG)                     // 8 stages per timestep
#define NBLK (G4H / BN)                        // 16 col blocks
#define NWG_PER_SEQ ((Bq / BM) * NBLK)         // 2*16 = 32
#define NWG (NSEQ * NWG_PER_SEQ)               // 96

// LDS: resident weight slice [BN][KCAT] f16 (256KB) + double-buffered A [BM][KSTG] (2x32KB)
#define SW_ELEMS ((size_t)BN * KCAT)
#define SA_ELEMS ((size_t)BM * KSTG)
#define LDS_BYTES (SW_ELEMS * 2 + 2 * SA_ELEMS * 2)   // 320KB == per-WG architectural max

// ---------------- workspace layout (bytes) ----------------
static constexpr size_t alignup(size_t x) { return (x + 255) & ~(size_t)255; }
static constexpr size_t SLOT_E   = (size_t)(Tq + 1) * Bq * Hq;       // per-seq elems (T+1 slots)
static constexpr size_t XBUF_B   = alignup((size_t)NSEQ * SLOT_E * sizeof(_Float16));
static constexpr size_t OFF_WT   = 0;
static constexpr size_t OFF_BIAS = alignup(OFF_WT + (size_t)Lq * G4H * KCAT * sizeof(_Float16));
static constexpr size_t OFF_XA   = alignup(OFF_BIAS + (size_t)Lq * G4H * sizeof(float));
static constexpr size_t OFF_XB   = OFF_XA + XBUF_B;
static constexpr size_t OFF_R    = OFF_XB + XBUF_B;
static constexpr size_t OFF_RN   = alignup(OFF_R  + (size_t)Bq * R3H * sizeof(float));
static constexpr size_t OFF_H1   = alignup(OFF_RN + (size_t)Bq * R3H * sizeof(float));
static constexpr size_t OFF_H1N  = alignup(OFF_H1 + (size_t)Bq * HDIM2 * sizeof(float));
static constexpr size_t OFF_H2   = alignup(OFF_H1N + (size_t)Bq * HDIM2 * sizeof(float));
static constexpr size_t OFF_H2N  = alignup(OFF_H2 + (size_t)Bq * HDIM3 * sizeof(float));
static constexpr size_t OFF_SYNC = alignup(OFF_H2N + (size_t)Bq * HDIM3 * sizeof(float));

// ---------------- helpers ----------------
__device__ inline float fast_rcp(float x) { return __builtin_amdgcn_rcpf(x); }
__device__ inline float fast_sigmoid(float x) { return fast_rcp(1.0f + __expf(-x)); }
__device__ inline float fast_tanh(float x) { return 1.0f - 2.0f * fast_rcp(1.0f + __expf(2.0f * x)); }

#if USE_ASYNC_LDS
__device__ __forceinline__ void async_copy16(const _Float16* g, _Float16* l) {
  __builtin_amdgcn_global_load_async_to_lds_b128((ASG v4i_t*)g, (ASL v4i_t*)l, 0, 0);
}
#endif

// LDS bank-conflict swizzles; `col` is a multiple of 8 f16 (one 16B granule).
// A stage tile: 16 granules/row, row stride 256B (= 64 banks) -> rotate by row&15.
__device__ inline int swzA(int row, int col) {
  int g = (((col >> 3) + row) & 15);
  return row * KSTG + g * 8;
}
// W slice: 128 granules/row, row stride 2KB (bank-aligned) -> rotate by row.
__device__ inline int swzB(int row, int col) {
  int g = (((col >> 3) + row) & 127);
  return row * KCAT + g * 8;
}

__device__ inline v16h ld_frag2(const _Float16* p0, const _Float16* p1) {
  union { uint4 u[2]; v16h h; } t;
  t.u[0] = *(const uint4*)p0;
  t.u[1] = *(const uint4*)p1;
  return t.h;
}

// ---------------- init: zero slot0 of both X buffers + sync counters ----------------
__global__ __launch_bounds__(256) void init_zero_k(_Float16* XA, _Float16* XB, int* syncw) {
  size_t total = (size_t)NSEQ * Bq * Hq;
  size_t stride = (size_t)gridDim.x * blockDim.x;
  for (size_t idx = (size_t)blockIdx.x * blockDim.x + threadIdx.x; idx < total; idx += stride) {
    size_t s = idx / ((size_t)Bq * Hq);
    size_t r = idx % ((size_t)Bq * Hq);
    XA[s * SLOT_E + r] = (_Float16)0.0f;
    XB[s * SLOT_E + r] = (_Float16)0.0f;
  }
  size_t g = (size_t)blockIdx.x * blockDim.x + threadIdx.x;
  if (g < 16) syncw[g] = 0;
}

// ---------------- weight convert: concat [Wx;Wh], transpose to [N',K], gate-interleave ----------------
// col'(q, j) = (j/16)*64 + q*16 + (j%16)   (q = gate, j = hidden unit)
__global__ __launch_bounds__(256) void cvt_w_k(const float* __restrict__ Wx,
                                               const float* __restrict__ Wh,
                                               const float* __restrict__ bsrc,
                                               _Float16* __restrict__ WT,
                                               float* __restrict__ biasp) {
  size_t total = (size_t)Lq * G4H * KCAT;
  size_t stride = (size_t)gridDim.x * blockDim.x;
  for (size_t idx = (size_t)blockIdx.x * blockDim.x + threadIdx.x; idx < total; idx += stride) {
    int l  = (int)(idx / ((size_t)G4H * KCAT));
    int r  = (int)(idx % ((size_t)G4H * KCAT));
    int np = r / KCAT;
    int k  = r % KCAT;
    int q  = (np >> 4) & 3;
    int j  = ((np >> 6) << 4) | (np & 15);
    int c  = q * Hq + j;
    float v = (k < Hq)
        ? Wx[(size_t)l * Hq * G4H + (size_t)k * G4H + c]
        : Wh[(size_t)l * Hq * G4H + (size_t)(k - Hq) * G4H + c];
    WT[idx] = (_Float16)v;
  }
  size_t btot = (size_t)Lq * G4H;
  for (size_t idx = (size_t)blockIdx.x * blockDim.x + threadIdx.x; idx < btot; idx += stride) {
    int l  = (int)(idx / G4H);
    int np = (int)(idx % G4H);
    int q  = (np >> 4) & 3;
    int j  = ((np >> 6) << 4) | (np & 15);
    biasp[idx] = bsrc[(size_t)l * G4H + q * Hq + j];
  }
}

// ---------------- embedding gather (f32 -> f16), write into slot t+1 ----------------
__global__ __launch_bounds__(256) void embed_k(const int* __restrict__ t1,
                                               const int* __restrict__ t2,
                                               const int* __restrict__ t3,
                                               const float* __restrict__ emb,
                                               _Float16* __restrict__ XA) {
  size_t total = (size_t)NSEQ * Tq * Bq * Hq;
  size_t stride = (size_t)gridDim.x * blockDim.x;
  for (size_t idx = (size_t)blockIdx.x * blockDim.x + threadIdx.x; idx < total; idx += stride) {
    int s    = (int)(idx / ((size_t)Tq * Bq * Hq));
    size_t r = idx % ((size_t)Tq * Bq * Hq);
    int t    = (int)(r / ((size_t)Bq * Hq));
    int r2   = (int)(r % ((size_t)Bq * Hq));
    int bb   = r2 / Hq;
    int hh   = r2 % Hq;
    const int* txt = (s == 0) ? t1 : ((s == 1) ? t2 : t3);
    int tok = txt[bb * Tq + t];
    float v = emb[(size_t)tok * Hq + hh];
    XA[(size_t)s * SLOT_E + (size_t)(t + 1) * Bq * Hq + r2] = (_Float16)v;
  }
}

// ---------------- persistent fused LSTM layer (all 3 sequences) ----------------
// LDS-resident weight slice; double-buffered async K=128 A staging; wave-local gates.
__global__ __launch_bounds__(256) void lstm_layer_k(const _Float16* __restrict__ Xin,
                                                    _Float16* __restrict__ Xout,
                                                    const _Float16* __restrict__ WT,   // [G4H][KCAT]
                                                    const float* __restrict__ biasp,   // [G4H]
                                                    int* __restrict__ sync_cnt) {
  extern __shared__ char smem_raw[];
  _Float16* sW = (_Float16*)smem_raw;                       // [BN][KCAT] swizzled
  _Float16* sA = (_Float16*)(smem_raw + SW_ELEMS * 2);      // 2 x [BM][KSTG] swizzled

  const int tid  = threadIdx.x;
  const int wg   = blockIdx.x;
  const int s    = wg / NWG_PER_SEQ;
  const int rem  = wg % NWG_PER_SEQ;
  const int m0   = (rem / NBLK) * BM;   // batch-row block (0 / 128)
  const int n0   = (rem % NBLK) * BN;   // permuted gate-col block

  const int wid  = tid >> 5;
  const int lane = tid & 31;
  const int wm   = wid >> 1;   // 0..3 : 32-row group
  const int wn   = wid & 1;    // 0..1 : 64-col group (4 gates x 16 units)
  const int lr   = lane & 15;
  const int hi   = lane >> 4;

  const _Float16* Xin_s  = Xin  + (size_t)s * SLOT_E;
  _Float16*       Xout_s = Xout + (size_t)s * SLOT_E;

  // ---- preload this WG's weight slice into LDS once (stays for all 256 steps)
  {
    const uint4* src = (const uint4*)(WT + (size_t)n0 * KCAT);
    uint4* dst = (uint4*)sW;
    for (int i = tid; i < (int)(SW_ELEMS / 8); i += 256) {
      int row = i >> 7;
      int gp  = ((i & 127) + row) & 127;
      dst[row * 128 + gp] = src[i];
    }
  }

  // this wave's 16 hidden units
  const int unit = (((n0 + wn * 64) >> 6) << 4) + lr;

  float bgate[4];
#pragma unroll
  for (int cc = 0; cc < 4; ++cc) bgate[cc] = biasp[n0 + wn * 64 + cc * 16 + lr];

  v8f cstate[2];
#pragma unroll
  for (int rr = 0; rr < 2; ++rr)
#pragma unroll
    for (int e = 0; e < 8; ++e) cstate[rr][e] = 0.0f;

  const int r_ = tid >> 1;        // A staging: row 0..127
  const int q_ = tid & 1;         // which 8-granule octet of the 16-granule row

  __syncthreads();                // weights resident

  for (int t = 0; t < Tq; ++t) {
    const _Float16* xsrc = Xin_s  + (size_t)(t + 1) * Bq * Hq;   // x_t
    const _Float16* hsrc = Xout_s + (size_t)t * Bq * Hq;         // h_{t-1} (slot0 zeros)

    v8f acc[2][4];
#pragma unroll
    for (int rr = 0; rr < 2; ++rr)
#pragma unroll
      for (int cc = 0; cc < 4; ++cc)
#pragma unroll
        for (int e = 0; e < 8; ++e) acc[rr][cc][e] = bgate[cc];

#if !USE_ASYNC_LDS
    uint4 pf[8];
#endif

    // ---- stage 0 (K [0,128), pure x-half)
    {
      const _Float16* g = xsrc + (size_t)(m0 + r_) * Hq + q_ * 64;
#pragma unroll
      for (int i = 0; i < 8; ++i) {
        _Float16* d = sA + swzA(r_, q_ * 64 + i * 8);
#if USE_ASYNC_LDS
        async_copy16(g + i * 8, d);
#else
        pf[i] = ((const uint4*)g)[i];
        *(uint4*)d = pf[i];
#endif
      }
#if USE_ASYNC_LDS
      __builtin_amdgcn_s_wait_asynccnt(0);
#endif
    }
    __syncthreads();

    for (int st = 0; st < NSTG; ++st) {
      // issue next stage's loads (overlap with the 32 WMMAs below)
      if (st < NSTG - 1) {
        const int kn = (st + 1) * KSTG;
        const _Float16* a = (kn < Hq) ? xsrc : hsrc;
        const int koff = (kn < Hq) ? kn : (kn - Hq);
        const _Float16* g = a + (size_t)(m0 + r_) * Hq + koff + q_ * 64;
#if USE_ASYNC_LDS
        _Float16* base = sA + ((st + 1) & 1) * SA_ELEMS;
#pragma unroll
        for (int i = 0; i < 8; ++i)
          async_copy16(g + i * 8, base + swzA(r_, q_ * 64 + i * 8));
#else
#pragma unroll
        for (int i = 0; i < 8; ++i) pf[i] = ((const uint4*)g)[i];
#endif
      }

      // ---- 4 WMMA-K chunks on the current staged tile
      const _Float16* abuf = sA + (st & 1) * SA_ELEMS;
#pragma unroll
      for (int kc2 = 0; kc2 < KSTG / 32; ++kc2) {
        const int kloc = kc2 * 32;             // col within stage tile
        const int k    = st * KSTG + kloc;     // absolute K for weight slice

        v16h af[2], bf[4];
#pragma unroll
        for (int rr = 0; rr < 2; ++rr) {
          int row = wm * 32 + rr * 16 + lr;
          af[rr] = ld_frag2(abuf + swzA(row, kloc + hi * 8),
                            abuf + swzA(row, kloc + hi * 8 + 16));
        }
#pragma unroll
        for (int cc = 0; cc < 4; ++cc) {
          int row = wn * 64 + cc * 16 + lr;
          bf[cc] = ld_frag2(sW + swzB(row, k + hi * 8),
                            sW + swzB(row, k + hi * 8 + 16));
        }
#pragma unroll
        for (int rr = 0; rr < 2; ++rr)
#pragma unroll
          for (int cc = 0; cc < 4; ++cc)
            acc[rr][cc] = __builtin_amdgcn_wmma_f32_16x16x32_f16(
                false, af[rr], false, bf[cc], (short)0, acc[rr][cc], false, false);
      }

      // commit next stage + single barrier per stage
      if (st < NSTG - 1) {
#if USE_ASYNC_LDS
        __builtin_amdgcn_s_wait_asynccnt(0);
#else
        _Float16* base = sA + ((st + 1) & 1) * SA_ELEMS;
#pragma unroll
        for (int i = 0; i < 8; ++i)
          *(uint4*)(base + swzA(r_, q_ * 64 + i * 8)) = pf[i];
#endif
        __syncthreads();
      }
    }

    // ---- wave-local LSTM nonlinearity (cc == gate i,f,g,o for same 16 units)
    _Float16* hdst = Xout_s + (size_t)(t + 1) * Bq * Hq;
#pragma unroll
    for (int rr = 0; rr < 2; ++rr) {
#pragma unroll
      for (int e = 0; e < 8; ++e) {
        float c = fast_sigmoid(acc[rr][1][e]) * cstate[rr][e] +
                  fast_sigmoid(acc[rr][0][e]) * fast_tanh(acc[rr][2][e]);
        cstate[rr][e] = c;
        float h = fast_sigmoid(acc[rr][3][e]) * fast_tanh(c);
        int row = m0 + wm * 32 + rr * 16 + hi * 8 + e;
        hdst[(size_t)row * Hq + unit] = (_Float16)h;
      }
    }

    // ---- device-wide barrier: all 96 WGs publish h_t before t+1
    __syncthreads();
    if (tid == 0) {
      __threadfence();
      atomicAdd(sync_cnt, 1);
      const int target = NWG * (t + 1);
      while (__hip_atomic_load(sync_cnt, __ATOMIC_RELAXED, __HIP_MEMORY_SCOPE_AGENT) < target) {
        __builtin_amdgcn_s_sleep(2);
      }
      __threadfence();
    }
    __syncthreads();
  }
}

// ---------------- r = concat(h_T of the 3 sequences) ----------------
__global__ __launch_bounds__(256) void gather_r_k(const _Float16* __restrict__ Xtop, float* __restrict__ R) {
  size_t total = (size_t)Bq * R3H;
  size_t stride = (size_t)gridDim.x * blockDim.x;
  for (size_t idx = (size_t)blockIdx.x * blockDim.x + threadIdx.x; idx < total; idx += stride) {
    int b = (int)(idx / R3H);
    int c = (int)(idx % R3H);
    int s = c / Hq;
    int j = c % Hq;
    R[idx] = (float)Xtop[(size_t)s * SLOT_E + (size_t)Tq * Bq * Hq + (size_t)b * Hq + j];
  }
}

// ---------------- batchnorm over batch dim (rows == 256 == blockDim) ----------------
__global__ __launch_bounds__(256) void bn_k(const float* __restrict__ in, float* __restrict__ out,
                                            int ncols, const float* __restrict__ gamma,
                                            const float* __restrict__ beta) {
  __shared__ float red[256];
  const int col = blockIdx.x;
  const int tid = threadIdx.x;
  float x = in[(size_t)tid * ncols + col];
  red[tid] = x;
  __syncthreads();
  for (int o = 128; o > 0; o >>= 1) { if (tid < o) red[tid] += red[tid + o]; __syncthreads(); }
  float mean = red[0] * (1.0f / 256.0f);
  __syncthreads();
  float d = x - mean;
  red[tid] = d * d;
  __syncthreads();
  for (int o = 128; o > 0; o >>= 1) { if (tid < o) red[tid] += red[tid + o]; __syncthreads(); }
  float var = red[0] * (1.0f / 256.0f);
  out[(size_t)tid * ncols + col] = gamma[col] * d * rsqrtf(var + 1e-3f) + beta[col];
}

// ---------------- small dense layer: out = act(A @ W + bias) ----------------
__global__ __launch_bounds__(256) void dense_k(const float* __restrict__ A, const float* __restrict__ W,
                                               const float* __restrict__ bias, float* __restrict__ out,
                                               int M, int N, int K, int selu) {
  const int tx = threadIdx.x & 15, ty = threadIdx.x >> 4;
  const int n = blockIdx.x * 16 + tx;
  const int m = blockIdx.y * 16 + ty;
  if (m >= M || n >= N) return;
  float acc = 0.0f;
  for (int k = 0; k < K; ++k) acc += A[(size_t)m * K + k] * W[(size_t)k * N + n];
  acc += bias[n];
  if (selu) {
    const float lam = 1.0507009873554805f, al = 1.6732632423543772f;
    acc = (acc > 0.0f) ? lam * acc : lam * al * (__expf(acc) - 1.0f);
  }
  out[(size_t)m * N + n] = acc;
}

// ---------------- host-side orchestration ----------------
extern "C" void kernel_launch(void* const* d_in, const int* in_sizes, int n_in,
                              void* d_out, int out_size, void* d_ws, size_t ws_size,
                              hipStream_t stream) {
  (void)in_sizes; (void)n_in; (void)out_size; (void)ws_size;
  const int*   text1 = (const int*)d_in[0];
  const int*   text2 = (const int*)d_in[1];
  const int*   text3 = (const int*)d_in[2];
  const float* emb   = (const float*)d_in[3];
  const float* Wx    = (const float*)d_in[4];
  const float* Wh    = (const float*)d_in[5];
  const float* blstm = (const float*)d_in[6];
  const float* g1    = (const float*)d_in[7];
  const float* beta1 = (const float*)d_in[8];
  const float* W1    = (const float*)d_in[9];
  const float* bd1   = (const float*)d_in[10];
  const float* g2    = (const float*)d_in[11];
  const float* beta2 = (const float*)d_in[12];
  const float* W2    = (const float*)d_in[13];
  const float* bd2   = (const float*)d_in[14];
  const float* g3    = (const float*)d_in[15];
  const float* beta3 = (const float*)d_in[16];
  const float* W3    = (const float*)d_in[17];
  const float* bd3   = (const float*)d_in[18];

  char* ws = (char*)d_ws;
  _Float16* WT    = (_Float16*)(ws + OFF_WT);
  float*    biasp = (float*)   (ws + OFF_BIAS);
  _Float16* XA    = (_Float16*)(ws + OFF_XA);
  _Float16* XB    = (_Float16*)(ws + OFF_XB);
  float*    Rb    = (float*)   (ws + OFF_R);
  float*    Rn    = (float*)   (ws + OFF_RN);
  float*    H1    = (float*)   (ws + OFF_H1);
  float*    H1n   = (float*)   (ws + OFF_H1N);
  float*    H2    = (float*)   (ws + OFF_H2);
  float*    H2n   = (float*)   (ws + OFF_H2N);
  int*      syncw = (int*)     (ws + OFF_SYNC);

  // allow 320KB dynamic LDS for the persistent LSTM kernel
  (void)hipFuncSetAttribute((const void*)lstm_layer_k,
                            hipFuncAttributeMaxDynamicSharedMemorySize,
                            (int)LDS_BYTES);

  // 1) zero h-slot0 + sync counters
  init_zero_k<<<1536, 256, 0, stream>>>(XA, XB, syncw);

  // 2) concatenated / transposed / gate-interleaved f16 weights + permuted bias
  cvt_w_k<<<8192, 256, 0, stream>>>(Wx, Wh, blstm, WT, biasp);

  // 3) embedding gather into layer-0 input (XA slots 1..T)
  embed_k<<<393216, 256, 0, stream>>>(text1, text2, text3, emb, XA);

  // 4) three LSTM layers, all sequences in parallel, persistent-kernel recurrence
  const size_t WSTRIDE = (size_t)G4H * KCAT;
  lstm_layer_k<<<NWG, 256, LDS_BYTES, stream>>>(XA, XB, WT + 0 * WSTRIDE, biasp + 0 * G4H, syncw + 0);
  lstm_layer_k<<<NWG, 256, LDS_BYTES, stream>>>(XB, XA, WT + 1 * WSTRIDE, biasp + 1 * G4H, syncw + 1);
  lstm_layer_k<<<NWG, 256, LDS_BYTES, stream>>>(XA, XB, WT + 2 * WSTRIDE, biasp + 2 * G4H, syncw + 2);

  // 5) r = concat(h_T)  (top layer output lives in XB, slot T)
  gather_r_k<<<1536, 256, 0, stream>>>(XB, Rb);

  // 6) head: BN -> SELU dense -> BN -> SELU dense -> BN -> dense
  bn_k<<<R3H, 256, 0, stream>>>(Rb, Rn, R3H, g1, beta1);
  dense_k<<<dim3((HDIM2 + 15) / 16, Bq / 16), 256, 0, stream>>>(Rn, W1, bd1, H1, Bq, HDIM2, R3H, 1);
  bn_k<<<HDIM2, 256, 0, stream>>>(H1, H1n, HDIM2, g2, beta2);
  dense_k<<<dim3((HDIM3 + 15) / 16, Bq / 16), 256, 0, stream>>>(H1n, W2, bd2, H2, Bq, HDIM3, HDIM2, 1);
  bn_k<<<HDIM3, 256, 0, stream>>>(H2, H2n, HDIM3, g3, beta3);
  dense_k<<<dim3(1, Bq / 16), 256, 0, stream>>>(H2n, W3, bd3, (float*)d_out, Bq, 4, HDIM3, 0);
}
